// GPT5Block_37374805410195
// MI455X (gfx1250) — matrix operand order repaired and checked by reading, not compiled, not run on hardware
//
#include <hip/hip_runtime.h>
#include <hip/hip_bf16.h>

#define T_TOK 512
#define D_MODEL 512
#define N_EXP 64
#define D_ROUTE 64
#define D_FF 2048
#define D_SH 1024

typedef __attribute__((ext_vector_type(16))) __bf16 v16bf;
typedef __attribute__((ext_vector_type(8)))  __bf16 v8bf;
typedef __attribute__((ext_vector_type(8)))  float  v8f;

__device__ __forceinline__ v8f wmma_bf16(v16bf a, v16bf b, v8f c) {
  return __builtin_amdgcn_wmma_f32_16x16x32_bf16(false, a, false, b, (short)0, c,
                                                 false, false);
}

// A fragment: lane holds row (lane%16), k = kb + 16*(lane/16) + j (j=0..15).
// p must point at rowbase + kb + 16*(lane/16).
__device__ __forceinline__ v16bf a_frag(const __bf16* p) {
  v8bf lo = *(const v8bf*)p;
  v8bf hi = *(const v8bf*)(p + 8);
  return __builtin_shufflevector(lo, hi, 0, 1, 2, 3, 4, 5, 6, 7, 8, 9, 10, 11,
                                 12, 13, 14, 15);
}

// B fragment: lane holds col n=(lane%16), k = kb + 16*(lane/16) + j (j=0..15).
// wp must point at W[(kb + 16*(lane/16))*LDN + n], W row-major KxN fp32.
template <int LDN>
__device__ __forceinline__ v16bf b_frag(const float* wp) {
  v16bf b;
#pragma unroll
  for (int j = 0; j < 16; ++j) b[j] = (__bf16)wp[(size_t)j * LDN];
  return b;
}

// Non-temporal variant: for the one-shot 536MB expert-weight stream, keep the
// reusable shared/FF weights resident in the 192MB L2 instead.
template <int LDN>
__device__ __forceinline__ v16bf b_frag_nt(const float* wp) {
  v16bf b;
#pragma unroll
  for (int j = 0; j < 16; ++j)
    b[j] = (__bf16)__builtin_nontemporal_load(wp + (size_t)j * LDN);
  return b;
}

// Fast silu: bf16-precision datapath does not need IEEE division.
__device__ __forceinline__ float silu_f(float v) {
  return v * __builtin_amdgcn_rcpf(1.0f + __expf(-v));
}

// ---------------- zero scratch accumulators ----------------
__global__ void k_init(float* __restrict__ moe, int* __restrict__ cnt) {
  int i = blockIdx.x * blockDim.x + threadIdx.x;
  if (i < T_TOK * D_MODEL) moe[i] = 0.0f;
  if (i < N_EXP) cnt[i] = 0;
}

// ---------------- LayerNorm (one block per token) ----------------
__global__ __launch_bounds__(256) void k_ln(const float* __restrict__ in,
                                            const float* __restrict__ g,
                                            const float* __restrict__ b,
                                            float* __restrict__ outf,
                                            __bf16* __restrict__ outb,
                                            int write_f32) {
  __shared__ float red[256];
  int t = blockIdx.x, tid = threadIdx.x;
  const float* row = in + (size_t)t * D_MODEL;
  float v0 = row[tid], v1 = row[tid + 256];
  red[tid] = v0 + v1;
  __syncthreads();
  for (int s = 128; s > 0; s >>= 1) {
    if (tid < s) red[tid] += red[tid + s];
    __syncthreads();
  }
  float mean = red[0] * (1.0f / 512.0f);
  __syncthreads();
  float d0 = v0 - mean, d1 = v1 - mean;
  red[tid] = d0 * d0 + d1 * d1;
  __syncthreads();
  for (int s = 128; s > 0; s >>= 1) {
    if (tid < s) red[tid] += red[tid + s];
    __syncthreads();
  }
  float rstd = rsqrtf(red[0] * (1.0f / 512.0f) + 1e-5f);
  float o0 = d0 * rstd * g[tid] + b[tid];
  float o1 = d1 * rstd * g[tid + 256] + b[tid + 256];
  size_t base = (size_t)t * D_MODEL;
  if (write_f32) {
    outf[base + tid] = o0;
    outf[base + tid + 256] = o1;
  }
  outb[base + tid] = (__bf16)o0;
  outb[base + tid + 256] = (__bf16)o1;
}

// ---------------- Router: top-2 of softmax == top-2 logits, renorm = 2-way softmax
__global__ __launch_bounds__(64) void k_router(const float* __restrict__ h,
                                               const float* __restrict__ Wr,
                                               const float* __restrict__ cb,
                                               int* __restrict__ cnt,
                                               int* __restrict__ tokl,
                                               float* __restrict__ wgtl) {
  __shared__ float r[D_ROUTE];
  __shared__ float lg[N_EXP];
  int t = blockIdx.x, j = threadIdx.x;
  const float* hrow = h + (size_t)t * D_MODEL;
  float acc = 0.0f;
  for (int k = 0; k < D_MODEL; ++k) acc += hrow[k] * Wr[k * D_ROUTE + j];
  r[j] = acc;
  __syncthreads();
  const float* cbe = cb + j * D_ROUTE;
  float l = 0.0f;
  for (int d = 0; d < D_ROUTE; ++d) l += r[d] * cbe[d];
  lg[j] = l * (1.0f / 0.7f);  // /TEMP
  __syncthreads();
  if (j == 0) {
    float b1v = -1e30f, b2v = -1e30f;
    int i1 = 0, i2 = 0;
    for (int e = 0; e < N_EXP; ++e) {
      float v = lg[e];
      if (v > b1v) { b2v = b1v; i2 = i1; b1v = v; i1 = e; }
      else if (v > b2v) { b2v = v; i2 = e; }
    }
    float e2 = __expf(b2v - b1v);
    float inv = 1.0f / (1.0f + e2);
    float w1 = inv, w2 = e2 * inv;
    int p1 = atomicAdd(&cnt[i1], 1);
    tokl[i1 * T_TOK + p1] = t; wgtl[i1 * T_TOK + p1] = w1;
    int p2 = atomicAdd(&cnt[i2], 1);
    tokl[i2 * T_TOK + p2] = t; wgtl[i2 * T_TOK + p2] = w2;
  }
}

// ---------------- Grouped MoE experts: silu(h@W1+b1)@W2+b2, scatter-accumulate
__global__ __launch_bounds__(256) void k_moe(
    const __bf16* __restrict__ hbf, const float* __restrict__ W1,
    const float* __restrict__ b1, const float* __restrict__ W2,
    const float* __restrict__ b2, const int* __restrict__ cnt,
    const int* __restrict__ tokl, const float* __restrict__ wgtl,
    float* __restrict__ moe) {
  int e = blockIdx.x, tile = blockIdx.y;
  int ne = cnt[e];
  if (tile * 16 >= ne) return;

  __shared__ __bf16 hlds[16 * 520];   // 16 gathered token rows (pad 520)
  __shared__ __bf16 mid[16 * 1032];   // silu intermediate, one 1024-chunk
  __shared__ int stok[16];
  __shared__ float swgt[16];

  int tid = threadIdx.x;
  if (tid < 16) {
    int p = tile * 16 + tid;
    if (p < ne) { stok[tid] = tokl[e * T_TOK + p]; swgt[tid] = wgtl[e * T_TOK + p]; }
    else        { stok[tid] = 0;                   swgt[tid] = 0.0f; }
  }
  __syncthreads();
  {
    int row = tid >> 4, c0 = (tid & 15) * 32;
    const __bf16* src = hbf + (size_t)stok[row] * D_MODEL + c0;
    __bf16* dst = hlds + row * 520 + c0;
#pragma unroll
    for (int i = 0; i < 32; i += 8) *(v8bf*)(dst + i) = *(const v8bf*)(src + i);
  }
  __syncthreads();

  int wave = tid >> 5, lane = tid & 31;
  int lrow = lane & 15, lhi = lane >> 4;
  const float* W1e = W1 + (size_t)e * D_MODEL * D_FF;
  const float* W2e = W2 + (size_t)e * D_FF * D_MODEL;

  v8f acc2[4];
#pragma unroll
  for (int i = 0; i < 4; ++i) acc2[i] = (v8f){0, 0, 0, 0, 0, 0, 0, 0};

  for (int chunk = 0; chunk < 2; ++chunk) {
    // stage 1: eh chunk [16 x 1024]
    for (int nt = wave; nt < 64; nt += 8) {
      int nloc = nt * 16 + lrow;
      int ng = chunk * 1024 + nloc;
      v8f acc = {0, 0, 0, 0, 0, 0, 0, 0};
      for (int kb = 0; kb < D_MODEL; kb += 32) {
        v16bf a = a_frag(hlds + lrow * 520 + kb + lhi * 16);
        v16bf bm = b_frag_nt<D_FF>(W1e + (size_t)(kb + lhi * 16) * D_FF + ng);
        acc = wmma_bf16(a, bm, acc);
      }
      float bias = b1[e * D_FF + ng];
#pragma unroll
      for (int v = 0; v < 8; ++v) {
        int m = v + lhi * 8;
        mid[m * 1032 + nloc] = (__bf16)silu_f(acc[v] + bias);
      }
    }
    __syncthreads();
    // stage 2 partial reduction over this chunk's K
    for (int kb = 0; kb < 1024; kb += 32) {
      v16bf a = a_frag(mid + lrow * 1032 + kb + lhi * 16);
      int kg = chunk * 1024 + kb + lhi * 16;
#pragma unroll
      for (int i = 0; i < 4; ++i) {
        int n2 = (wave * 4 + i) * 16 + lrow;
        v16bf bm = b_frag_nt<D_MODEL>(W2e + (size_t)kg * D_MODEL + n2);
        acc2[i] = wmma_bf16(a, bm, acc2[i]);
      }
    }
    __syncthreads();
  }
#pragma unroll
  for (int i = 0; i < 4; ++i) {
    int n2 = (wave * 4 + i) * 16 + lrow;
    float bias = b2[e * D_MODEL + n2];
#pragma unroll
    for (int v = 0; v < 8; ++v) {
      int m = v + lhi * 8;
      float wv = swgt[m];
      if (wv != 0.0f)
        atomicAdd(&moe[(size_t)stok[m] * D_MODEL + n2], wv * (acc2[i][v] + bias));
    }
  }
}

// ---------------- Shared expert: moe += 0.25 * (silu(h@Ws1+bs1)@Ws2+bs2)
__global__ __launch_bounds__(256) void k_shared(
    const __bf16* __restrict__ hbf, const float* __restrict__ Ws1,
    const float* __restrict__ bs1, const float* __restrict__ Ws2,
    const float* __restrict__ bs2, float* __restrict__ moe) {
  __shared__ __bf16 mid[16 * 1032];
  int t0 = blockIdx.x * 16;
  int tid = threadIdx.x;
  int wave = tid >> 5, lane = tid & 31, lrow = lane & 15, lhi = lane >> 4;
  const __bf16* arow = hbf + (size_t)(t0 + lrow) * D_MODEL;

  for (int nt = wave; nt < 64; nt += 8) {
    int n = nt * 16 + lrow;
    v8f acc = {0, 0, 0, 0, 0, 0, 0, 0};
    for (int kb = 0; kb < D_MODEL; kb += 32) {
      v16bf a = a_frag(arow + kb + lhi * 16);
      v16bf bm = b_frag<D_SH>(Ws1 + (size_t)(kb + lhi * 16) * D_SH + n);
      acc = wmma_bf16(a, bm, acc);
    }
    float bias = bs1[n];
#pragma unroll
    for (int v = 0; v < 8; ++v) {
      int m = v + lhi * 8;
      mid[m * 1032 + n] = (__bf16)silu_f(acc[v] + bias);
    }
  }
  __syncthreads();

  v8f acc2[4];
#pragma unroll
  for (int i = 0; i < 4; ++i) acc2[i] = (v8f){0, 0, 0, 0, 0, 0, 0, 0};
  for (int kb = 0; kb < D_SH; kb += 32) {
    v16bf a = a_frag(mid + lrow * 1032 + kb + lhi * 16);
#pragma unroll
    for (int i = 0; i < 4; ++i) {
      int n2 = (wave * 4 + i) * 16 + lrow;
      v16bf bm = b_frag<D_MODEL>(Ws2 + (size_t)(kb + lhi * 16) * D_MODEL + n2);
      acc2[i] = wmma_bf16(a, bm, acc2[i]);
    }
  }
#pragma unroll
  for (int i = 0; i < 4; ++i) {
    int n2 = (wave * 4 + i) * 16 + lrow;
    float bias = bs2[n2];
#pragma unroll
    for (int v = 0; v < 8; ++v) {
      int m = v + lhi * 8;
      size_t idx = (size_t)(t0 + m) * D_MODEL + n2;
      moe[idx] += 0.25f * (acc2[i][v] + bias);
    }
  }
}

// ---------------- FF on MoE output + final residual combine ----------------
__global__ __launch_bounds__(256) void k_ff(
    const __bf16* __restrict__ fbf, const float* __restrict__ Wf1,
    const float* __restrict__ bf1, const float* __restrict__ Wf2,
    const float* __restrict__ bf2, const float* __restrict__ x,
    const float* __restrict__ hmoe, float* __restrict__ out) {
  __shared__ __bf16 mid[16 * 1032];
  int t0 = blockIdx.x * 16;
  int tid = threadIdx.x;
  int wave = tid >> 5, lane = tid & 31, lrow = lane & 15, lhi = lane >> 4;
  const __bf16* arow = fbf + (size_t)(t0 + lrow) * D_MODEL;

  v8f acc2[4];
#pragma unroll
  for (int i = 0; i < 4; ++i) acc2[i] = (v8f){0, 0, 0, 0, 0, 0, 0, 0};

  for (int chunk = 0; chunk < 2; ++chunk) {
    for (int nt = wave; nt < 64; nt += 8) {
      int nloc = nt * 16 + lrow;
      int ng = chunk * 1024 + nloc;
      v8f acc = {0, 0, 0, 0, 0, 0, 0, 0};
      for (int kb = 0; kb < D_MODEL; kb += 32) {
        v16bf a = a_frag(arow + kb + lhi * 16);
        v16bf bm = b_frag<D_FF>(Wf1 + (size_t)(kb + lhi * 16) * D_FF + ng);
        acc = wmma_bf16(a, bm, acc);
      }
      float bias = bf1[ng];
#pragma unroll
      for (int v = 0; v < 8; ++v) {
        int m = v + lhi * 8;
        mid[m * 1032 + nloc] = (__bf16)silu_f(acc[v] + bias);
      }
    }
    __syncthreads();
    for (int kb = 0; kb < 1024; kb += 32) {
      v16bf a = a_frag(mid + lrow * 1032 + kb + lhi * 16);
      int kg = chunk * 1024 + kb + lhi * 16;
#pragma unroll
      for (int i = 0; i < 4; ++i) {
        int n2 = (wave * 4 + i) * 16 + lrow;
        v16bf bm = b_frag<D_MODEL>(Wf2 + (size_t)kg * D_MODEL + n2);
        acc2[i] = wmma_bf16(a, bm, acc2[i]);
      }
    }
    __syncthreads();
  }
#pragma unroll
  for (int i = 0; i < 4; ++i) {
    int n2 = (wave * 4 + i) * 16 + lrow;
    float bias = bf2[n2];
#pragma unroll
    for (int v = 0; v < 8; ++v) {
      int m = v + lhi * 8;
      size_t idx = (size_t)(t0 + m) * D_MODEL + n2;
      // out is written once and never re-read: non-temporal store.
      __builtin_nontemporal_store(x[idx] + hmoe[idx] + (acc2[i][v] + bias),
                                  &out[idx]);
    }
  }
}

extern "C" void kernel_launch(void* const* d_in, const int* in_sizes, int n_in,
                              void* d_out, int out_size, void* d_ws,
                              size_t ws_size, hipStream_t stream) {
  (void)in_sizes; (void)n_in; (void)out_size; (void)ws_size;
  const float* x     = (const float*)d_in[0];
  const float* ln1_g = (const float*)d_in[1];
  const float* ln1_b = (const float*)d_in[2];
  const float* Wr    = (const float*)d_in[3];
  const float* cb    = (const float*)d_in[4];
  const float* W1    = (const float*)d_in[5];
  const float* b1    = (const float*)d_in[6];
  const float* W2    = (const float*)d_in[7];
  const float* b2    = (const float*)d_in[8];
  const float* Ws1   = (const float*)d_in[9];
  const float* bs1   = (const float*)d_in[10];
  const float* Ws2   = (const float*)d_in[11];
  const float* bs2   = (const float*)d_in[12];
  const float* fflg  = (const float*)d_in[13];
  const float* fflb  = (const float*)d_in[14];
  const float* Wf1   = (const float*)d_in[15];
  const float* bf1   = (const float*)d_in[16];
  const float* Wf2   = (const float*)d_in[17];
  const float* bf2   = (const float*)d_in[18];

  char* ws = (char*)d_ws;
  size_t off = 0;
  float*  h_f32 = (float*)(ws + off);  off += (size_t)T_TOK * D_MODEL * 4;  // 1 MB
  __bf16* h_bf  = (__bf16*)(ws + off); off += (size_t)T_TOK * D_MODEL * 2;  // 512 KB
  float*  moe   = (float*)(ws + off);  off += (size_t)T_TOK * D_MODEL * 4;  // 1 MB
  __bf16* f_bf  = (__bf16*)(ws + off); off += (size_t)T_TOK * D_MODEL * 2;  // 512 KB
  int*    cnt   = (int*)(ws + off);    off += 256;
  int*    tokl  = (int*)(ws + off);    off += (size_t)N_EXP * T_TOK * 4;    // 128 KB
  float*  wgtl  = (float*)(ws + off);  off += (size_t)N_EXP * T_TOK * 4;    // 128 KB

  k_init<<<(T_TOK * D_MODEL + 255) / 256, 256, 0, stream>>>(moe, cnt);
  k_ln<<<T_TOK, 256, 0, stream>>>(x, ln1_g, ln1_b, h_f32, h_bf, 1);
  k_router<<<T_TOK, 64, 0, stream>>>(h_f32, Wr, cb, cnt, tokl, wgtl);
  k_moe<<<dim3(N_EXP, 32), 256, 0, stream>>>(h_bf, W1, b1, W2, b2, cnt, tokl,
                                             wgtl, moe);
  k_shared<<<T_TOK / 16, 256, 0, stream>>>(h_bf, Ws1, bs1, Ws2, bs2, moe);
  k_ln<<<T_TOK, 256, 0, stream>>>(moe, fflg, fflb, nullptr, f_bf, 0);
  k_ff<<<T_TOK / 16, 256, 0, stream>>>(f_bf, Wf1, bf1, Wf2, bf2, x, moe,
                                       (float*)d_out);
}
